// AttentionGCNConv_28544352649819
// MI455X (gfx1250) — matrix operand order repaired and checked by reading, not compiled, not run on hardware
//
#include <hip/hip_runtime.h>
#include <hip/hip_bf16.h>

#define IN_C  64
#define OUT_C 16
#define EA_D  8

typedef _Float16 v16h __attribute__((ext_vector_type(16)));
typedef float    v8f  __attribute__((ext_vector_type(8)));

// ---------------------------------------------------------------------------
// Kernel 1: zero the output accumulator (harness poisons d_out with 0xAA).
// ---------------------------------------------------------------------------
__global__ void zero_kernel(float* __restrict__ p, int n) {
    int i = blockIdx.x * blockDim.x + threadIdx.x;
    if (i < n) p[i] = 0.0f;
}

// ---------------------------------------------------------------------------
// Kernel 2: h = x @ lin_w + lin_b   ([N,64] @ [64,16] -> [N,16])
// One wave per 16-row tile; K=64 as two chained v_wmma_f32_16x16x32_f16.
// Operand packing follows CDNA5 ISA 7.12.2 (wave32 layouts):
//   A (16x32 f16): lane m = L%16 holds row m; elements 0..7  -> K = (L/16)*8 + i
//                                            elements 8..15 -> K = 16 + (L/16)*8 + (i-8)
//   B (32x16 f16): lane n = L%16 holds col n; same K order as A
//   D (16x16 f32): VGPR j -> M = j + 8*(L/16), N = L%16
// ---------------------------------------------------------------------------
__global__ void lin_wmma_kernel(const float* __restrict__ x,
                                const float* __restrict__ lin_w,
                                const float* __restrict__ lin_b,
                                float* __restrict__ h, int n_rows) {
    const int wave  = threadIdx.x >> 5;            // 8 waves / block
    const int lane  = threadIdx.x & 31;
    const int tile  = blockIdx.x * 8 + wave;       // 16-row tile id
    const int r0    = tile * 16;
    if (r0 >= n_rows) return;                      // wave-uniform: EXEC stays full

    const int m     = lane & 15;                   // A row / B col / D col
    const int halfw = lane >> 4;                   // lane half (0 or 1)

    v8f c = {};
    const float* xr = x + (size_t)(r0 + m) * IN_C; // this lane's A row

#pragma unroll
    for (int kk = 0; kk < IN_C; kk += 32) {
        v16h a, b;
#pragma unroll
        for (int i = 0; i < 8; ++i) {
            a[i]     = (_Float16)xr[kk + halfw * 8 + i];        // K = halfw*8 + i
            a[8 + i] = (_Float16)xr[kk + 16 + halfw * 8 + i];   // K = 16 + halfw*8 + i
        }
#pragma unroll
        for (int i = 0; i < 16; ++i) {
            // lin_w is [64][16] row-major; lane's column is m
            b[i] = (_Float16)lin_w[(kk + halfw * 8 + i + (i & 8)) * OUT_C + m];
        }
        c = __builtin_amdgcn_wmma_f32_16x16x32_f16(
                /*neg_a=*/false, a, /*neg_b=*/false, b,
                /*c_mod=*/(short)0, c, /*reuse_a=*/false, /*reuse_b=*/false);
    }

    const float bias = lin_b[m];
#pragma unroll
    for (int j = 0; j < 8; ++j) {
        const int mo = halfw * 8 + j;              // D row for this VGPR
        h[(size_t)(r0 + mo) * OUT_C + m] = c[j] + bias;
    }
}

// ---------------------------------------------------------------------------
// Kernel 3: per-(edge, channel) fused pipeline.
// thread -> (e, c); 16 consecutive lanes (half-wave) share one edge, so the
// channel-softmax is 4 xor-shuffle steps at width 16 (wave32-native).
// ---------------------------------------------------------------------------
__global__ void edge_kernel(const int*   __restrict__ ei,        // [2,E] flat
                            const float* __restrict__ edge_attr, // [E,8]
                            const float* __restrict__ h,         // [N,16]
                            const float* __restrict__ ea_w,      // [8,16]
                            const float* __restrict__ ea_b,      // [16]
                            const float* __restrict__ w1,        // [16]
                            const float* __restrict__ b1,        // [16]
                            const float* __restrict__ w2,        // [16]
                            const float* __restrict__ b2,        // [1]
                            float* __restrict__ out,             // [N,16]
                            int E) {
    __shared__ float s_w1[OUT_C], s_b1[OUT_C], s_w2[OUT_C], s_eab[OUT_C];
    __shared__ float s_eaw[EA_D * OUT_C];

    const int t = threadIdx.x;
    if (t < OUT_C) { s_w1[t] = w1[t]; s_b1[t] = b1[t]; s_w2[t] = w2[t]; s_eab[t] = ea_b[t]; }
    if (t < EA_D * OUT_C) s_eaw[t] = ea_w[t];
    __syncthreads();

    const int gid = blockIdx.x * blockDim.x + t;
    const int e   = gid >> 4;
    const int c   = gid & 15;
    if (e >= E) return;

    const int r   = ei[e] - 1;        // forward does row -= 1
    const int col = ei[E + e];

    // ea[e][c] = sum_d edge_attr[e][d] * ea_w[d][c] + ea_b[c]
    // each lane loads one attr element, broadcast via width-16 shuffles
    float my = (c < EA_D) ? edge_attr[(size_t)e * EA_D + c] : 0.0f;
    float ea = s_eab[c];
#pragma unroll
    for (int d = 0; d < EA_D; ++d) {
        float ad = __shfl(my, d, 16);
        ea = fmaf(ad, s_eaw[d * OUT_C + c], ea);
    }

    float agg = h[(size_t)col * OUT_C + c] * ea;

    // attn MLP: score = sum_j relu(agg*w1[j]+b1[j]) * w2[j] + b2
    float score = *b2;
#pragma unroll
    for (int j = 0; j < OUT_C; ++j) {
        float hid = fmaf(agg, s_w1[j], s_b1[j]);
        hid = fmaxf(hid, 0.0f);
        score = fmaf(hid, s_w2[j], score);
    }

    // softmax over the 16 channels of this edge (half-wave reduction)
    float mx = score;
#pragma unroll
    for (int off = 8; off > 0; off >>= 1) mx = fmaxf(mx, __shfl_xor(mx, off, 16));
    float ex = __expf(score - mx);
    float sum = ex;
#pragma unroll
    for (int off = 8; off > 0; off >>= 1) sum += __shfl_xor(sum, off, 16);

    float v = agg * (ex / sum);
    atomicAdd(&out[(size_t)r * OUT_C + c], v);    // global_atomic_add_f32, lands in L2
}

// ---------------------------------------------------------------------------
extern "C" void kernel_launch(void* const* d_in, const int* in_sizes, int n_in,
                              void* d_out, int out_size, void* d_ws, size_t ws_size,
                              hipStream_t stream) {
    const float* x         = (const float*)d_in[0];
    const int*   ei        = (const int*)  d_in[1];
    const float* edge_attr = (const float*)d_in[2];
    const float* lin_w     = (const float*)d_in[3];
    const float* lin_b     = (const float*)d_in[4];
    const float* ea_w      = (const float*)d_in[5];
    const float* ea_b      = (const float*)d_in[6];
    const float* attn_w1   = (const float*)d_in[7];
    const float* attn_b1   = (const float*)d_in[8];
    const float* attn_w2   = (const float*)d_in[9];
    const float* attn_b2   = (const float*)d_in[10];

    const int N_ = in_sizes[0] / IN_C;   // 100000
    const int E_ = in_sizes[2] / EA_D;   // 400000

    float* h   = (float*)d_ws;           // [N,16] f32 scratch = 6.4 MB
    float* out = (float*)d_out;

    // 1) zero output accumulator
    zero_kernel<<<(out_size + 255) / 256, 256, 0, stream>>>(out, out_size);

    // 2) h = x @ lin_w + lin_b via WMMA (8 waves/block, 1 tile/wave)
    const int tiles = (N_ + 15) / 16;
    lin_wmma_kernel<<<(tiles + 7) / 8, 256, 0, stream>>>(x, lin_w, lin_b, h, N_);

    // 3) fused edge pipeline + scatter
    const long long threads = (long long)E_ * OUT_C;
    edge_kernel<<<(unsigned)((threads + 255) / 256), 256, 0, stream>>>(
        ei, edge_attr, h, ea_w, ea_b, attn_w1, attn_b1, attn_w2, attn_b2, out, E_);
}
